// GraphConv_72086731096858
// MI455X (gfx1250) — matrix operand order repaired and checked by reading, not compiled, not run on hardware
//
#include <hip/hip_runtime.h>

#define N_NODES 100000
#define N_EDGES 1600000
#define IN_FEATS 256
#define OUT_FEATS 128

typedef unsigned short ushort_t;
typedef __attribute__((ext_vector_type(8)))  float    v8f;
typedef __attribute__((ext_vector_type(8)))  unsigned int v8u;
typedef __attribute__((ext_vector_type(16))) ushort_t v16us;
typedef __attribute__((ext_vector_type(16))) __bf16   v16bf;

// ---------------------------------------------------------------------------
// Kernel 1: zero degree arrays (2*N floats, contiguous in ws) and the output.
// ---------------------------------------------------------------------------
__global__ void gcn_zero_kernel(float* __restrict__ deg, float* __restrict__ out) {
    const long ndeg  = 2L * N_NODES;
    const long nout  = (long)N_NODES * OUT_FEATS;
    const long total = ndeg + nout;
    for (long i = blockIdx.x * (long)blockDim.x + threadIdx.x; i < total;
         i += (long)gridDim.x * blockDim.x) {
        if (i < ndeg) deg[i] = 0.0f;
        else          out[i - ndeg] = 0.0f;
    }
}

// ---------------------------------------------------------------------------
// Kernel 2: per-edge degree accumulation (float atomics are exact for counts).
// ---------------------------------------------------------------------------
__global__ void gcn_deg_kernel(const int* __restrict__ src, const int* __restrict__ dst,
                               float* __restrict__ deg_src, float* __restrict__ deg_dst) {
    int e = blockIdx.x * blockDim.x + threadIdx.x;
    if (e < N_EDGES) {
        atomicAdd(&deg_src[src[e]], 1.0f);
        atomicAdd(&deg_dst[dst[e]], 1.0f);
    }
}

// ---------------------------------------------------------------------------
// Kernel 3: split W into bf16 hi/lo and transpose: Wt_*[c][k], c=out col.
// A row of Wt_* is then exactly the contiguous-K run a B fragment lane needs.
// ---------------------------------------------------------------------------
__global__ void gcn_wprep_kernel(const float* __restrict__ W,
                                 ushort_t* __restrict__ Wt_h,
                                 ushort_t* __restrict__ Wt_l) {
    int i = blockIdx.x * blockDim.x + threadIdx.x;   // over 256*128
    if (i < IN_FEATS * OUT_FEATS) {
        int k = i >> 7;            // 0..255 (input feature)
        int c = i & 127;           // 0..127 (output col)
        float x = W[i];
        unsigned int b = __builtin_bit_cast(unsigned int, x);
        ushort_t hi = (ushort_t)(b >> 16);                       // truncate to bf16
        float r = x - __builtin_bit_cast(float, b & 0xFFFF0000u);
        ushort_t lo = (ushort_t)(__builtin_bit_cast(unsigned int, r) >> 16);
        Wt_h[c * IN_FEATS + k] = hi;
        Wt_l[c * IN_FEATS + k] = lo;
    }
}

// ---------------------------------------------------------------------------
// Kernel 4: h = (feat * norm_src) @ W via split-bf16 V_WMMA_F32_16X16X32_BF16
// (3 products: AhBh + AhBl + AlBh, f32 accumulate ~= fp32 GEMM).
// Block = 256 threads = 8 waves; wave owns one 16-row M tile x full N=128
// (8 v8f accumulators). W hi/lo K-slice (16 KB) staged in LDS, fragments
// read with aligned ds_load_b128 pairs (no register shuffling).
// ---------------------------------------------------------------------------
__launch_bounds__(256)
__global__ void gcn_gemm_kernel(const float* __restrict__ feat,
                                const ushort_t* __restrict__ Wt_h,
                                const ushort_t* __restrict__ Wt_l,
                                const float* __restrict__ deg_src,
                                float* __restrict__ h) {
    // Lds[c][0..31] = Wh K-slice, Lds[c][32..63] = Wl K-slice ; 128*64*2B = 16 KB
    __shared__ ushort_t Lds[128 * 64];

    const int tid  = threadIdx.x;
    const int lane = tid & 31;
    const int wave = tid >> 5;
    const int col  = lane & 15;         // N (B/C) or M (A) sub-index
    const int half = lane >> 4;         // lane half selects K sub-range
    const int rowbase = (blockIdx.x * 8 + wave) * 16;

    int rowA = rowbase + col;
    if (rowA >= N_NODES) rowA = N_NODES - 1;    // clamp; OOB rows never stored
    float d = deg_src[rowA];
    const float nsrc = rsqrtf(d < 1.0f ? 1.0f : d);
    const float* __restrict__ arow = feat + (long)rowA * IN_FEATS;

    v8f acc[8] = {};

    for (int k0 = 0; k0 < IN_FEATS; k0 += 32) {
        __syncthreads();
        // Stage 128x32 hi + lo W slice: 1024 16-byte chunks, 4 per thread.
#pragma unroll
        for (int i = 0; i < 4; ++i) {
            int m = tid + i * 256;          // chunk id 0..1023
            int c = m >> 3;                 // 0..127
            int q = (m & 7) * 8;            // 0,8,...,56 (ushort index)
            const ushort_t* sp = (q < 32) ? (Wt_h + c * IN_FEATS + k0 + q)
                                          : (Wt_l + c * IN_FEATS + k0 + (q - 32));
            *(uint4*)(Lds + c * 64 + q) = *(const uint4*)sp;
        }
        __syncthreads();

        // A fragment: elements 0-15 = K 16h..16h+15 contiguous (two 32B runs)
        v8f r0 = *(const v8f*)(arow + k0 + 16 * half);
        v8f r1 = *(const v8f*)(arow + k0 + 16 * half + 8);
        v16us ahu, alu;
#pragma unroll
        for (int i = 0; i < 8; ++i) {
            float x0 = r0[i] * nsrc;
            unsigned int b0 = __builtin_bit_cast(unsigned int, x0);
            ahu[i] = (ushort_t)(b0 >> 16);
            float rr0 = x0 - __builtin_bit_cast(float, b0 & 0xFFFF0000u);
            alu[i] = (ushort_t)(__builtin_bit_cast(unsigned int, rr0) >> 16);

            float x1 = r1[i] * nsrc;
            unsigned int b1 = __builtin_bit_cast(unsigned int, x1);
            ahu[8 + i] = (ushort_t)(b1 >> 16);
            float rr1 = x1 - __builtin_bit_cast(float, b1 & 0xFFFF0000u);
            alu[8 + i] = (ushort_t)(__builtin_bit_cast(unsigned int, rr1) >> 16);
        }
        v16bf ah = __builtin_bit_cast(v16bf, ahu);
        v16bf al = __builtin_bit_cast(v16bf, alu);

#pragma unroll
        for (int j = 0; j < 8; ++j) {
            // B fragment lane: col = N, needs K 16h..16h+15 contiguous (32B)
            const ushort_t* bp = Lds + (j * 16 + col) * 64 + 16 * half;
            v16bf bh = __builtin_bit_cast(v16bf, *(const v8u*)(bp));
            v16bf bl = __builtin_bit_cast(v16bf, *(const v8u*)(bp + 32));
            acc[j] = __builtin_amdgcn_wmma_f32_16x16x32_bf16(
                false, ah, false, bh, (short)0, acc[j], false, false);
            acc[j] = __builtin_amdgcn_wmma_f32_16x16x32_bf16(
                false, ah, false, bl, (short)0, acc[j], false, false);
            acc[j] = __builtin_amdgcn_wmma_f32_16x16x32_bf16(
                false, al, false, bh, (short)0, acc[j], false, false);
        }
    }

    // C/D layout: VGPR r -> M = r (lanes 0-15) / r+8 (lanes 16-31), N = col.
    const int mofs = half * 8;
#pragma unroll
    for (int j = 0; j < 8; ++j) {
#pragma unroll
        for (int r = 0; r < 8; ++r) {
            int grow = rowbase + r + mofs;
            if (grow < N_NODES)
                h[(long)grow * OUT_FEATS + j * 16 + col] = acc[j][r];
        }
    }
}

// ---------------------------------------------------------------------------
// Kernel 5: SpMM scatter. One wave32 per edge: float4 gather of h[src],
// 4x global_atomic_add_f32 per lane into out[dst]. h and out are both
// L2-resident (51 MB each vs 192 MB L2).
// ---------------------------------------------------------------------------
__global__ void gcn_scatter_kernel(const int* __restrict__ src, const int* __restrict__ dst,
                                   const float* __restrict__ h, float* __restrict__ out) {
    long gid  = blockIdx.x * (long)blockDim.x + threadIdx.x;
    long e    = gid >> 5;
    int  lane = (int)(gid & 31);
    if (e < N_EDGES) {
        int s = src[e];
        int d = dst[e];
        const float4* hv = (const float4*)(h + (long)s * OUT_FEATS);
        float4 v = hv[lane];
        float* o = out + (long)d * OUT_FEATS + lane * 4;
        atomicAdd(o + 0, v.x);
        atomicAdd(o + 1, v.y);
        atomicAdd(o + 2, v.z);
        atomicAdd(o + 3, v.w);
    }
}

// ---------------------------------------------------------------------------
// Kernel 6: out = out * norm_dst + bias
// ---------------------------------------------------------------------------
__global__ void gcn_final_kernel(const float* __restrict__ deg_dst,
                                 const float* __restrict__ bias,
                                 float* __restrict__ out) {
    long i = blockIdx.x * (long)blockDim.x + threadIdx.x;
    if (i < (long)N_NODES * OUT_FEATS) {
        int n = (int)(i >> 7);
        int c = (int)(i & 127);
        float d = deg_dst[n];
        float nd = rsqrtf(d < 1.0f ? 1.0f : d);
        out[i] = out[i] * nd + bias[c];
    }
}

// ---------------------------------------------------------------------------
extern "C" void kernel_launch(void* const* d_in, const int* in_sizes, int n_in,
                              void* d_out, int out_size, void* d_ws, size_t ws_size,
                              hipStream_t stream) {
    const float* feat = (const float*)d_in[0];
    const int*   src  = (const int*)d_in[1];
    const int*   dst  = (const int*)d_in[2];
    const float* W    = (const float*)d_in[3];
    const float* bias = (const float*)d_in[4];
    float* out = (float*)d_out;

    // ws layout: deg_src[N] | deg_dst[N] | h[N*128] f32 | Wt_h | Wt_l (bf16)
    float* deg_src = (float*)d_ws;
    float* deg_dst = deg_src + N_NODES;
    float* h       = deg_dst + N_NODES;
    ushort_t* Wt_h = (ushort_t*)(h + (long)N_NODES * OUT_FEATS);
    ushort_t* Wt_l = Wt_h + IN_FEATS * OUT_FEATS;

    gcn_zero_kernel<<<2048, 256, 0, stream>>>(deg_src, out);

    gcn_deg_kernel<<<(N_EDGES + 255) / 256, 256, 0, stream>>>(src, dst, deg_src, deg_dst);

    gcn_wprep_kernel<<<(IN_FEATS * OUT_FEATS + 255) / 256, 256, 0, stream>>>(W, Wt_h, Wt_l);

    const int mtiles = (N_NODES + 15) / 16;                    // 6250
    gcn_gemm_kernel<<<(mtiles + 7) / 8, 256, 0, stream>>>(feat, Wt_h, Wt_l, deg_src, h);

    gcn_scatter_kernel<<<(int)(((long)N_EDGES * 32 + 255) / 256), 256, 0, stream>>>(src, dst, h, out);

    gcn_final_kernel<<<(int)(((long)N_NODES * OUT_FEATS + 255) / 256), 256, 0, stream>>>(deg_dst, bias, out);
}
